// SmolVLMWithExpertModel_38534446580090
// MI455X (gfx1250) — compile-verified, matrix operand order checked by reading, not run on hardware
//
#include <hip/hip_runtime.h>
#include <hip/hip_bf16.h>

// ---------------- model dims ----------------
#define BB     4
#define L0C    560
#define L1C    51
#define LC     (L0C + L1C)      // 611
#define HV     576
#define HE     512
#define HEADS  9
#define KVH    3
#define HD     64
#define INTC   1536
#define NLAY   18
#define EPSF   1e-6f

typedef __attribute__((ext_vector_type(16))) __bf16 v16bf;
typedef __attribute__((ext_vector_type(8)))  float  v8f;

union FragBF { v16bf v; unsigned short u[16]; };

// native bf16 convert (gfx1250 has hardware bf16 cvt)
__device__ __forceinline__ unsigned short f2bf(float f) {
    union { __bf16 h; unsigned short s; } c;
    c.h = (__bf16)f;
    return c.s;
}

// =====================================================================
// Generic WMMA GEMM:  C[M,N] (+)= A[M,K] @ W[N,K]^T   (bf16 in, f32 acc)
// Double-buffered LDS pipeline: one barrier per K-step, next tile's
// global loads overlap current tile's WMMAs.
// A and C rows can be remapped through (seg, tok_stride, row_off) so the
// same kernel writes/reads the position-concatenated Q/K/V/AO buffers.
// =====================================================================
#define BM 64
#define BN 64
#define BK 32
#define LSTR 40   // LDS row stride (elems) to dodge bank conflicts

__global__ __launch_bounds__(256) void k_gemm(
    const float* __restrict__ A, int lda, int a_seg, int a_ts, int a_off,
    const float* __restrict__ W, int ldw,
    float* __restrict__ C, int ldc, int c_seg, int c_ts, int c_off,
    int M, int N, int K, int mode /*0=store 1=add*/)
{
    __shared__ unsigned short As[2][BM * LSTR];
    __shared__ unsigned short Ws[2][BN * LSTR];

    const int tid  = threadIdx.x;
    const int lane = tid & 31;
    const int wid  = tid >> 5;
    const int m0   = blockIdx.y * BM;
    const int n0   = blockIdx.x * BN;
    const int wm   = (wid >> 1) * 16;   // wave row base (0..48)
    const int wn   = (wid & 1) * 32;    // wave col base (0 or 32)

    v8f acc0 = {}; v8f acc1 = {};

    const int lrow = tid >> 2;          // 0..63
    const int lcol = (tid & 3) * 8;     // 0,8,16,24

    // precomputed source row pointers (row remap applied once)
    const float* aRowPtr = nullptr;
    {
        int r = m0 + lrow;
        if (r < M) {
            int b = r / a_seg; int t = r - b * a_seg;
            aRowPtr = A + (size_t)(b * a_ts + a_off + t) * lda + lcol;
        }
    }
    const float* wRowPtr = nullptr;
    {
        int n = n0 + lrow;
        if (n < N) wRowPtr = W + (size_t)n * ldw + lcol;
    }

    const int nk = K / BK;              // all K dims are multiples of 32

    auto fetch = [&](int kt, float* ra, float* rw) {
        const int k0 = kt * BK;
        if (aRowPtr) {
            const float* s = aRowPtr + k0;
            if (kt + 1 < nk) __builtin_prefetch(s + BK, 0, 3);
            #pragma unroll
            for (int i = 0; i < 8; i++) ra[i] = s[i];
        } else {
            #pragma unroll
            for (int i = 0; i < 8; i++) ra[i] = 0.f;
        }
        if (wRowPtr) {
            const float* s = wRowPtr + k0;
            if (kt + 1 < nk) __builtin_prefetch(s + BK, 0, 3);
            #pragma unroll
            for (int i = 0; i < 8; i++) rw[i] = s[i];
        } else {
            #pragma unroll
            for (int i = 0; i < 8; i++) rw[i] = 0.f;
        }
    };
    auto stage = [&](int buf, const float* ra, const float* rw) {
        #pragma unroll
        for (int i = 0; i < 8; i++) As[buf][lrow * LSTR + lcol + i] = f2bf(ra[i]);
        #pragma unroll
        for (int i = 0; i < 8; i++) Ws[buf][lrow * LSTR + lcol + i] = f2bf(rw[i]);
    };

    {
        float ra[8], rw[8];
        fetch(0, ra, rw);
        stage(0, ra, rw);
    }
    __syncthreads();

    const int lm = lane & 15, half = lane >> 4;
    for (int kt = 0; kt < nk; kt++) {
        const int cur = kt & 1;
        float ra[8], rw[8];
        const bool more = (kt + 1 < nk);
        if (more) fetch(kt + 1, ra, rw);   // global loads in flight during WMMA

        FragBF fa, fb0, fb1;
        #pragma unroll
        for (int j = 0; j < 16; j++) {
            int k = (j < 8) ? (half * 8 + j) : (16 + half * 8 + (j - 8));
            fa.u[j]  = As[cur][(wm + lm) * LSTR + k];
            fb0.u[j] = Ws[cur][(wn + lm) * LSTR + k];
            fb1.u[j] = Ws[cur][(wn + 16 + lm) * LSTR + k];
        }
        acc0 = __builtin_amdgcn_wmma_f32_16x16x32_bf16(false, fa.v, false, fb0.v,
                                                       (short)0, acc0, false, false);
        acc1 = __builtin_amdgcn_wmma_f32_16x16x32_bf16(false, fa.v, false, fb1.v,
                                                       (short)0, acc1, false, false);
        if (more) stage(1 - cur, ra, rw);
        __syncthreads();
    }

    // ---- epilogue: C/D layout m = i + 8*half, n = lane%16 ----
    #pragma unroll
    for (int i = 0; i < 8; i++) {
        int m = m0 + wm + i + 8 * half;
        if (m >= M) continue;
        int b = m / c_seg; int t = m - b * c_seg;
        size_t row = (size_t)(b * c_ts + c_off + t);
        int n1 = n0 + wn + lm;
        if (n1 < N) { float* p = C + row * ldc + n1; *p = mode ? (*p + acc0[i]) : acc0[i]; }
        int n2 = n1 + 16;
        if (n2 < N) { float* p = C + row * ldc + n2; *p = mode ? (*p + acc1[i]) : acc1[i]; }
    }
}

// =====================================================================
// Attention: one block (128 thr / 4 waves) per (b, head, 16-query tile)
// =====================================================================
#define LPAD 640   // 611 keys padded to 40 tiles of 16

__global__ __launch_bounds__(128) void k_attn(
    const float* __restrict__ Q, const float* __restrict__ Kb,
    const float* __restrict__ Vb, const float* __restrict__ bias,
    float* __restrict__ AO, float scale)
{
    __shared__ float          sc[16 * LPAD];      // scores then probs (f32)
    __shared__ unsigned short vt[64 * 34];        // V tile transposed (bf16)
    __shared__ float          red[16][8];

    const int q0  = blockIdx.x * 16;
    const int h   = blockIdx.y;
    const int b   = blockIdx.z;
    const int tid = threadIdx.x;
    const int lane = tid & 31, wid = tid >> 5;
    const int lm = lane & 15, half = lane >> 4;
    const int kvh = h / (HEADS / KVH);

    // ---- Q fragments (two K-halves of head dim 64) ----
    FragBF qa, qb;
    {
        int q = q0 + lm;
        const float* qp = (q < LC) ? Q + (size_t)(b * LC + q) * (HEADS * HD) + h * HD : nullptr;
        #pragma unroll
        for (int j = 0; j < 16; j++) {
            int k = (j < 8) ? (half * 8 + j) : (16 + half * 8 + (j - 8));
            qa.u[j] = qp ? f2bf(qp[k])      : 0;
            qb.u[j] = qp ? f2bf(qp[32 + k]) : 0;
        }
    }

    // ---- phase 1: S = Q @ K^T, 16x16 per key tile, waves strided ----
    const int nkt = (LC + 15) / 16;
    for (int kt = wid; kt < nkt; kt += 4) {
        FragBF ka, kb2;
        int key = kt * 16 + lm;
        const float* kp = (key < LC) ? Kb + (size_t)(b * LC + key) * (KVH * HD) + kvh * HD : nullptr;
        #pragma unroll
        for (int j = 0; j < 16; j++) {
            int k = (j < 8) ? (half * 8 + j) : (16 + half * 8 + (j - 8));
            ka.u[j]  = kp ? f2bf(kp[k])      : 0;
            kb2.u[j] = kp ? f2bf(kp[32 + k]) : 0;
        }
        v8f s = {};
        s = __builtin_amdgcn_wmma_f32_16x16x32_bf16(false, qa.v, false, ka.v,  (short)0, s, false, false);
        s = __builtin_amdgcn_wmma_f32_16x16x32_bf16(false, qb.v, false, kb2.v, (short)0, s, false, false);
        #pragma unroll
        for (int i = 0; i < 8; i++)
            sc[(i + 8 * half) * LPAD + kt * 16 + lm] = s[i];
    }
    __syncthreads();

    // ---- phase 2: softmax(scale*S + bias) per row ----
    const int row = tid & 15, sub = tid >> 4;   // 8 threads per row
    const int q = q0 + row;
    float mx = -1e30f;
    if (q < LC) {
        const float* bp = bias + ((size_t)b * LC + q) * LC;
        for (int c = sub; c < LC; c += 8) {
            float v = sc[row * LPAD + c] * scale + bp[c];
            sc[row * LPAD + c] = v;
            mx = fmaxf(mx, v);
        }
    }
    red[row][sub] = mx; __syncthreads();
    mx = red[row][0];
    #pragma unroll
    for (int i = 1; i < 8; i++) mx = fmaxf(mx, red[row][i]);
    __syncthreads();
    float sum = 0.f;
    if (q < LC) {
        for (int c = sub; c < LC; c += 8) {
            float e = __expf(sc[row * LPAD + c] - mx);
            sc[row * LPAD + c] = e;
            sum += e;
        }
    }
    red[row][sub] = sum; __syncthreads();
    sum = 0.f;
    #pragma unroll
    for (int i = 0; i < 8; i++) sum += red[row][i];
    float inv = (sum > 0.f) ? 1.f / sum : 0.f;
    for (int c = sub; c < LPAD; c += 8) {
        float p = (q < LC && c < LC) ? sc[row * LPAD + c] * inv : 0.f;
        sc[row * LPAD + c] = p;
    }

    // ---- phase 3: O = P @ V (wave wid owns d-range wid*16..+15) ----
    v8f o = {};
    const int d0 = wid * 16;
    const int nch = (LC + 31) / 32;
    for (int kc = 0; kc < nch; kc++) {
        __syncthreads();
        for (int idx = tid; idx < 32 * 64; idx += 128) {
            int key = idx >> 6, d = idx & 63;
            int kg = kc * 32 + key;
            float v = (kg < LC) ? Vb[(size_t)(b * LC + kg) * (KVH * HD) + kvh * HD + d] : 0.f;
            vt[d * 34 + key] = f2bf(v);
        }
        __syncthreads();
        FragBF pa, va;
        #pragma unroll
        for (int j = 0; j < 16; j++) {
            int k = (j < 8) ? (half * 8 + j) : (16 + half * 8 + (j - 8));
            pa.u[j] = f2bf(sc[lm * LPAD + kc * 32 + k]);
            va.u[j] = vt[(d0 + lm) * 34 + k];
        }
        o = __builtin_amdgcn_wmma_f32_16x16x32_bf16(false, pa.v, false, va.v, (short)0, o, false, false);
    }
    #pragma unroll
    for (int i = 0; i < 8; i++) {
        int qq = q0 + i + 8 * half;
        if (qq < LC)
            AO[(size_t)(b * LC + qq) * (HEADS * HD) + h * HD + d0 + lm] = o[i];
    }
}

// =====================================================================
// Small elementwise / norm kernels
// =====================================================================
__global__ void k_rope_tables(const int* __restrict__ pos,
                              float* __restrict__ cosT, float* __restrict__ sinT)
{
    int idx = blockIdx.x * blockDim.x + threadIdx.x;
    if (idx >= LC * 32) return;
    int t = idx >> 5, i = idx & 31;
    float invf = __powf(10000.f, -(float)i / 32.f);
    float f = (float)pos[t] * invf;
    cosT[idx] = __cosf(f);
    sinT[idx] = __sinf(f);
}

// in-place RoPE on [B, L, nh, 64] buffer (pairs d, d+32)
__global__ void k_rope(float* __restrict__ X, const float* __restrict__ cosT,
                       const float* __restrict__ sinT, int nh, int total)
{
    int idx = blockIdx.x * blockDim.x + threadIdx.x;
    if (idx >= total) return;
    int i = idx & 31; int rest = idx >> 5;
    int hh = rest % nh; rest /= nh;
    int t = rest % LC;  int b = rest / LC;
    float c = cosT[t * 32 + i], s = sinT[t * 32 + i];
    float* p = X + (size_t)(b * LC + t) * (nh * HD) + hh * HD;
    float x1 = p[i], x2 = p[i + 32];
    p[i]      = x1 * c - x2 * s;
    p[i + 32] = x2 * c + x1 * s;
}

__global__ __launch_bounds__(256) void k_rmsnorm(
    const float* __restrict__ X, const float* __restrict__ w,
    float* __restrict__ Y, int width)
{
    int rowi = blockIdx.x;
    const float* x = X + (size_t)rowi * width;
    float* y = Y + (size_t)rowi * width;
    __shared__ float r[256];
    float s = 0.f;
    for (int i = threadIdx.x; i < width; i += 256) { float v = x[i]; s += v * v; }
    r[threadIdx.x] = s; __syncthreads();
    for (int st = 128; st > 0; st >>= 1) {
        if (threadIdx.x < st) r[threadIdx.x] += r[threadIdx.x + st];
        __syncthreads();
    }
    float inv = rsqrtf(r[0] / width + EPSF);
    for (int i = threadIdx.x; i < width; i += 256) y[i] = x[i] * inv * w[i];
}

// ss[b, 0:2HE] = cond[b] @ pw^T + pb   (tiny GEMV: B=4 rows)
__global__ void k_ada_ss(const float* __restrict__ cond, const float* __restrict__ pw,
                         const float* __restrict__ pb, float* __restrict__ ss)
{
    int j = blockIdx.x * blockDim.x + threadIdx.x;
    if (j >= BB * 2 * HE) return;
    int b = j / (2 * HE), jj = j - b * 2 * HE;
    const float* c = cond + b * HE;
    const float* w = pw + (size_t)jj * HE;
    float acc = pb[jj];
    for (int k = 0; k < HE; k++) acc += c[k] * w[k];
    ss[j] = acc;
}

// y = rmsnorm(x, nw) * (1 + scale[b]) + shift[b]
__global__ __launch_bounds__(256) void k_ada_apply(
    const float* __restrict__ X, const float* __restrict__ nw,
    const float* __restrict__ ss, float* __restrict__ Y)
{
    int rowi = blockIdx.x;             // rows = B*L1
    int b = rowi / L1C;
    const float* x = X + (size_t)rowi * HE;
    float* y = Y + (size_t)rowi * HE;
    const float* sc = ss + (size_t)b * 2 * HE;
    __shared__ float r[256];
    float s = 0.f;
    for (int i = threadIdx.x; i < HE; i += 256) { float v = x[i]; s += v * v; }
    r[threadIdx.x] = s; __syncthreads();
    for (int st = 128; st > 0; st >>= 1) {
        if (threadIdx.x < st) r[threadIdx.x] += r[threadIdx.x + st];
        __syncthreads();
    }
    float inv = rsqrtf(r[0] / HE + EPSF);
    for (int i = threadIdx.x; i < HE; i += 256)
        y[i] = x[i] * inv * nw[i] * (1.f + sc[i]) + sc[HE + i];
}

__global__ void k_silu_mul(float* __restrict__ g, const float* __restrict__ u, int n)
{
    int idx = blockIdx.x * blockDim.x + threadIdx.x;
    if (idx >= n) return;
    float x = g[idx];
    g[idx] = (x / (1.f + __expf(-x))) * u[idx];
}

// =====================================================================
// Host orchestration
// =====================================================================
extern "C" void kernel_launch(void* const* d_in, const int* in_sizes, int n_in,
                              void* d_out, int out_size, void* d_ws, size_t ws_size,
                              hipStream_t stream)
{
    const float* prefix_emb = (const float*)d_in[0];
    const float* suffix_emb = (const float*)d_in[1];
    const float* cond       = (const float*)d_in[2];
    const float* bias       = (const float*)d_in[3];
    const int*   pos        = (const int*)  d_in[4];
    const float* Pv[9];  for (int i = 0; i < 9;  i++) Pv[i] = (const float*)d_in[5 + i];
    const float* Pe[13]; for (int i = 0; i < 13; i++) Pe[i] = (const float*)d_in[14 + i];
    const float* vlm_norm = (const float*)d_in[27];
    const float* exp_norm = (const float*)d_in[28];

    float* W = (float*)d_ws;
    size_t off = 0;
    auto alloc = [&](size_t n) { float* p = W + off; off += n; return p; };
    float* cosT = alloc((size_t)LC * 32);
    float* sinT = alloc((size_t)LC * 32);
    float* ph = alloc((size_t)BB * L0C * HV);
    float* sh = alloc((size_t)BB * L1C * HE);
    float* pn = alloc((size_t)BB * L0C * HV);
    float* sn = alloc((size_t)BB * L1C * HE);
    float* ss = alloc((size_t)BB * 2 * HE);
    float* Qb = alloc((size_t)BB * LC * HEADS * HD);
    float* Kb = alloc((size_t)BB * LC * KVH * HD);
    float* Vb = alloc((size_t)BB * LC * KVH * HD);
    float* AO = alloc((size_t)BB * LC * HEADS * HD);
    float* Gb = alloc((size_t)BB * L0C * INTC);
    float* Ub = alloc((size_t)BB * L0C * INTC);

    const int Mp = BB * L0C;   // 2240 prefix tokens
    const int Ms = BB * L1C;   // 204 suffix tokens

    hipMemcpyAsync(ph, prefix_emb, sizeof(float) * (size_t)Mp * HV, hipMemcpyDeviceToDevice, stream);
    hipMemcpyAsync(sh, suffix_emb, sizeof(float) * (size_t)Ms * HE, hipMemcpyDeviceToDevice, stream);

    k_rope_tables<<<(LC * 32 + 255) / 256, 256, 0, stream>>>(pos, cosT, sinT);

    auto gemm = [&](const float* A, int lda, int aseg, int ats, int aoff,
                    const float* Wt, int ldw,
                    float* C, int ldc, int cseg, int cts, int coff,
                    int M, int N, int K, int mode) {
        dim3 g((N + BN - 1) / BN, (M + BM - 1) / BM);
        k_gemm<<<g, 256, 0, stream>>>(A, lda, aseg, ats, aoff, Wt, ldw,
                                      C, ldc, cseg, cts, coff, M, N, K, mode);
    };

    for (int l = 0; l < NLAY; l++) {
        const float* vqw = Pv[0] + (size_t)l * HEADS * HD * HV;
        const float* vkw = Pv[1] + (size_t)l * KVH * HD * HV;
        const float* vvw = Pv[2] + (size_t)l * KVH * HD * HV;
        const float* vow = Pv[3] + (size_t)l * HV * HEADS * HD;
        const float* vgw = Pv[4] + (size_t)l * INTC * HV;
        const float* vuw = Pv[5] + (size_t)l * INTC * HV;
        const float* vdw = Pv[6] + (size_t)l * HV * INTC;
        const float* vln1 = Pv[7] + (size_t)l * HV;
        const float* vln2 = Pv[8] + (size_t)l * HV;

        const float* eqw = Pe[0] + (size_t)l * HEADS * HD * HE;
        const float* ekw = Pe[1] + (size_t)l * KVH * HD * HE;
        const float* evw = Pe[2] + (size_t)l * KVH * HD * HE;
        const float* eow = Pe[3] + (size_t)l * HE * HEADS * HD;
        const float* egw = Pe[4] + (size_t)l * INTC * HE;
        const float* euw = Pe[5] + (size_t)l * INTC * HE;
        const float* edw = Pe[6] + (size_t)l * HE * INTC;
        const float* eln1n  = Pe[7]  + (size_t)l * HE;
        const float* eln1pw = Pe[8]  + (size_t)l * 2 * HE * HE;
        const float* eln1pb = Pe[9]  + (size_t)l * 2 * HE;
        const float* eln2n  = Pe[10] + (size_t)l * HE;
        const float* eln2pw = Pe[11] + (size_t)l * 2 * HE * HE;
        const float* eln2pb = Pe[12] + (size_t)l * 2 * HE;

        // ---- pre-attention norms ----
        k_rmsnorm<<<Mp, 256, 0, stream>>>(ph, vln1, pn, HV);
        k_ada_ss<<<(BB * 2 * HE + 255) / 256, 256, 0, stream>>>(cond, eln1pw, eln1pb, ss);
        k_ada_apply<<<Ms, 256, 0, stream>>>(sh, eln1n, ss, sn);

        // ---- QKV (writes into position-concatenated buffers) ----
        gemm(pn, HV, L0C, L0C, 0, vqw, HV, Qb, HEADS * HD, L0C, LC, 0,   Mp, HEADS * HD, HV, 0);
        gemm(pn, HV, L0C, L0C, 0, vkw, HV, Kb, KVH * HD,   L0C, LC, 0,   Mp, KVH * HD,   HV, 0);
        gemm(pn, HV, L0C, L0C, 0, vvw, HV, Vb, KVH * HD,   L0C, LC, 0,   Mp, KVH * HD,   HV, 0);
        gemm(sn, HE, L1C, L1C, 0, eqw, HE, Qb, HEADS * HD, L1C, LC, L0C, Ms, HEADS * HD, HE, 0);
        gemm(sn, HE, L1C, L1C, 0, ekw, HE, Kb, KVH * HD,   L1C, LC, L0C, Ms, KVH * HD,   HE, 0);
        gemm(sn, HE, L1C, L1C, 0, evw, HE, Vb, KVH * HD,   L1C, LC, L0C, Ms, KVH * HD,   HE, 0);

        // ---- RoPE ----
        k_rope<<<(BB * LC * HEADS * 32 + 255) / 256, 256, 0, stream>>>(Qb, cosT, sinT, HEADS, BB * LC * HEADS * 32);
        k_rope<<<(BB * LC * KVH   * 32 + 255) / 256, 256, 0, stream>>>(Kb, cosT, sinT, KVH,   BB * LC * KVH   * 32);

        // ---- attention ----
        {
            dim3 g((LC + 15) / 16, HEADS, BB);
            k_attn<<<g, 128, 0, stream>>>(Qb, Kb, Vb, bias, AO, 0.125f);
        }

        // ---- output projections (residual add) ----
        gemm(AO, HEADS * HD, L0C, LC, 0,   vow, HEADS * HD, ph, HV, L0C, L0C, 0, Mp, HV, HEADS * HD, 1);
        gemm(AO, HEADS * HD, L1C, LC, L0C, eow, HEADS * HD, sh, HE, L1C, L1C, 0, Ms, HE, HEADS * HD, 1);

        // ---- MLP prefix ----
        k_rmsnorm<<<Mp, 256, 0, stream>>>(ph, vln2, pn, HV);
        gemm(pn, HV, L0C, L0C, 0, vgw, HV, Gb, INTC, L0C, L0C, 0, Mp, INTC, HV, 0);
        gemm(pn, HV, L0C, L0C, 0, vuw, HV, Ub, INTC, L0C, L0C, 0, Mp, INTC, HV, 0);
        k_silu_mul<<<((size_t)Mp * INTC + 255) / 256, 256, 0, stream>>>(Gb, Ub, Mp * INTC);
        gemm(Gb, INTC, L0C, L0C, 0, vdw, INTC, ph, HV, L0C, L0C, 0, Mp, HV, INTC, 1);

        // ---- MLP suffix ----
        k_ada_ss<<<(BB * 2 * HE + 255) / 256, 256, 0, stream>>>(cond, eln2pw, eln2pb, ss);
        k_ada_apply<<<Ms, 256, 0, stream>>>(sh, eln2n, ss, sn);
        gemm(sn, HE, L1C, L1C, 0, egw, HE, Gb, INTC, L1C, L1C, 0, Ms, INTC, HE, 0);
        gemm(sn, HE, L1C, L1C, 0, euw, HE, Ub, INTC, L1C, L1C, 0, Ms, INTC, HE, 0);
        k_silu_mul<<<((size_t)Ms * INTC + 255) / 256, 256, 0, stream>>>(Gb, Ub, Ms * INTC);
        gemm(Gb, INTC, L1C, L1C, 0, edw, INTC, sh, HE, L1C, L1C, 0, Ms, HE, INTC, 1);
    }

    // ---- final norms straight into d_out (prefix then suffix) ----
    float* out = (float*)d_out;
    k_rmsnorm<<<Mp, 256, 0, stream>>>(ph, vlm_norm, out, HV);
    k_rmsnorm<<<Ms, 256, 0, stream>>>(sh, exp_norm, out + (size_t)Mp * HV, HE);
}
